// SelfAttention_53798760350183
// MI455X (gfx1250) — matrix operand order repaired and checked by reading, not compiled, not run on hardware
//
#include <hip/hip_runtime.h>
#include <hip/hip_bf16.h>

// ---------------------------------------------------------------------------
// Self-attention (B=2,S=2048,E=1024,H=16,HD=64) via CDNA5 WMMA f16.
//  k1: QKV = x @ Win^T + bin      (f32 in -> f16 qkv ws, q pre-scaled 1/8)
//  k2: flash attention over qkv   (f16 ws -> f16 attn ws)
//      - softmax denominator tracked as an extra WMMA accumulator (P @ ones)
//      - K tiles staged global->LDS with async-to-LDS (ASYNCcnt) path
//  k3: out = attn @ Wout^T + bout (f16 ws -> f32 d_out)
// ---------------------------------------------------------------------------

typedef __attribute__((ext_vector_type(16))) _Float16 v16h;
typedef __attribute__((ext_vector_type(8)))  _Float16 v8h;
typedef __attribute__((ext_vector_type(4)))  _Float16 v4h;
typedef __attribute__((ext_vector_type(8)))  float    v8f;

#define B_   2
#define S_   2048
#define E_   1024
#define H_   16
#define HD_  64
#define MROWS (B_ * S_)          // 4096
#define QKV_HALVES ((size_t)B_ * H_ * S_ * HD_)   // 4,194,304 per tensor

static __device__ __forceinline__ v8f wmma_f16(v16h a, v16h b, v8f c) {
    return __builtin_amdgcn_wmma_f32_16x16x32_f16(
        /*neg_a=*/false, a, /*neg_b=*/false, b,
        /*c_mod=*/(short)0, c, /*reuse_a=*/false, /*reuse_b=*/false);
}

// Build a v16h fragment from two 16-byte chunks (8 halves each).
static __device__ __forceinline__ v16h ld2(const _Float16* p0, const _Float16* p1) {
    v8h lo = *(const v8h*)p0;
    v8h hi = *(const v8h*)(p1);
    v16h r;
#pragma unroll
    for (int i = 0; i < 8; ++i) { r[i] = lo[i]; r[8 + i] = hi[i]; }
    return r;
}

// Async global->LDS 16B copy (CDNA5 ASYNCcnt path). LDS byte address is the
// low 32 bits of the generic pointer.
static __device__ __forceinline__ void async_copy_b128(const _Float16* lds_dst,
                                                       const _Float16* gsrc) {
    unsigned lds_off = (unsigned)(uintptr_t)lds_dst;
    asm volatile("global_load_async_to_lds_b128 %0, %1, off"
                 :: "v"(lds_off), "v"((const void*)gsrc)
                 : "memory");
}
static __device__ __forceinline__ void wait_async0() {
    asm volatile("s_wait_asynccnt 0x0" ::: "memory");
}

// ---------------------------------------------------------------------------
// GEMM: C[M,N] = A[M,K] @ W[N,K]^T + bias[N]
// Block: 256 threads = 8 waves. Block tile 128(M) x 32(N), K chunked by 32.
// MODE 0: epilogue scatters f16 into head-major qkv workspace (q scaled 1/8).
// MODE 1: epilogue writes f32 to d_out.
// ---------------------------------------------------------------------------
template <bool A_F32, int MODE>
__global__ __launch_bounds__(256) void gemm_wmma_kernel(
    const void* __restrict__ Aptr, const float* __restrict__ W,
    const float* __restrict__ bias, void* __restrict__ Out,
    int M, int N, int K)
{
    constexpr int ASTR = 40;   // halves per LDS row (80B, 16B multiple)
    constexpr int WSTR = 40;
    __shared__ _Float16 Alds[128 * ASTR];
    __shared__ _Float16 Wlds[32 * WSTR];

    const int tid    = threadIdx.x;
    const int wave   = tid >> 5;
    const int lane   = tid & 31;
    const int lane16 = lane & 15;
    const int hi     = (lane >> 4) & 1;
    const int bM     = blockIdx.x;   // M / 128
    const int bN     = blockIdx.y;   // N / 32

    const int arow_s = tid >> 1;            // A staging: 2 threads per row
    const int acb    = (tid & 1) * 16;
    const size_t agrow = (size_t)bM * 128 + arow_s;
    const int wnl  = tid >> 3;              // W staging: 8 threads per row
    const int wkbi = (tid & 7) * 4;
    const float* wrow = W + ((size_t)bN * 32 + wnl) * K;

    v8f acc0 = {}, acc1 = {};

    for (int kb = 0; kb < K; kb += 32) {
        // prefetch next chunk while this one is consumed
        if (kb + 32 < K) {
            if (A_F32)
                __builtin_prefetch((const float*)Aptr + agrow * K + kb + 32 + acb, 0, 1);
            else
                __builtin_prefetch((const _Float16*)Aptr + agrow * K + kb + 32 + acb, 0, 1);
            __builtin_prefetch(wrow + kb + 32 + wkbi, 0, 1);
        }
        __syncthreads();
        // ---- stage A: 128 rows x 32 k
        _Float16* adst = &Alds[arow_s * ASTR + acb];
        if (A_F32) {
            const float* ap = (const float*)Aptr + agrow * K + kb + acb;
            v8h o0, o1;
#pragma unroll
            for (int i = 0; i < 8; ++i) o0[i] = (_Float16)ap[i];
#pragma unroll
            for (int i = 0; i < 8; ++i) o1[i] = (_Float16)ap[8 + i];
            *(v8h*)adst       = o0;
            *(v8h*)(adst + 8) = o1;
        } else {
            const _Float16* ap = (const _Float16*)Aptr + agrow * K + kb + acb;
            async_copy_b128(adst,     ap);       // ASYNCcnt DMA, no VGPR bounce
            async_copy_b128(adst + 8, ap + 8);
        }
        // ---- stage W: 32 rows x 32 k (f32 -> f16 conversion)
        {
            const float* wp = wrow + kb + wkbi;
            v4h o;
#pragma unroll
            for (int i = 0; i < 4; ++i) o[i] = (_Float16)wp[i];
            *(v4h*)&Wlds[wnl * WSTR + wkbi] = o;
        }
        if (!A_F32) wait_async0();
        __syncthreads();

        // ---- fragments + 2 WMMA per wave
        const _Float16* arow = &Alds[(wave * 16 + lane16) * ASTR];
        v16h a = ld2(arow + (hi ? 16 : 0), arow + (hi ? 24 : 8));
        const _Float16* w0 = &Wlds[(0 * 16 + lane16) * WSTR + (hi ? 16 : 0)];
        const _Float16* w1 = &Wlds[(1 * 16 + lane16) * WSTR + (hi ? 16 : 0)];
        v16h b0 = ld2(w0, w0 + 8);
        v16h b1 = ld2(w1, w1 + 8);
        acc0 = wmma_f16(a, b0, acc0);
        acc1 = wmma_f16(a, b1, acc1);
    }

    // ---- epilogue -----------------------------------------------------------
#pragma unroll
    for (int t = 0; t < 2; ++t) {
        const v8f acc = t ? acc1 : acc0;
        const int gcol = bN * 32 + t * 16 + lane16;
        const float bv = bias[gcol];
        if (MODE == 0) {
            // scatter into f16 qkv workspace: [which][b][h][s][hd]
            _Float16* qkv = (_Float16*)Out;
            const int which = gcol >> 10;       // 0=q 1=k 2=v
            const int e  = gcol & 1023;
            const int h  = e >> 6;
            const int hd = e & 63;
            const float scale = (which == 0) ? 0.125f : 1.0f;  // 1/sqrt(64)
#pragma unroll
            for (int r = 0; r < 8; ++r) {
                const int grow = bM * 128 + wave * 16 + r + hi * 8;
                const int b = grow >> 11;        // / S_
                const int s = grow & 2047;
                const size_t di = (size_t)which * QKV_HALVES +
                                  (((size_t)b * H_ + h) * S_ + s) * HD_ + hd;
                qkv[di] = (_Float16)((acc[r] + bv) * scale);
            }
        } else {
            float* out = (float*)Out;
#pragma unroll
            for (int r = 0; r < 8; ++r) {
                const int grow = bM * 128 + wave * 16 + r + hi * 8;
                out[(size_t)grow * N + gcol] = acc[r] + bv;
            }
        }
    }
}

// ---------------------------------------------------------------------------
// Flash attention. Grid: (S/128, B*H). Block: 256 threads = 8 waves.
// Each wave owns 16 queries; block shares K/V chunks of 32 keys via LDS.
// Per iteration: 4 WMMA scores, online softmax (max via half-wave shuffles,
// denominator via an extra P@ones WMMA accumulator), 4 WMMA for P@V.
// ---------------------------------------------------------------------------
__global__ __launch_bounds__(256) void flash_attn_kernel(
    const _Float16* __restrict__ qkv, _Float16* __restrict__ attnout)
{
    constexpr int KSTR = 72;   // 144B row, 16B multiple
    constexpr int VSTR = 40;   // transposed V: [hd][key], 80B row
    constexpr int PSTR = 40;
    __shared__ _Float16 Klds[32 * KSTR];
    __shared__ _Float16 Vlds[64 * VSTR];
    __shared__ _Float16 Plds[8 * 16 * PSTR];

    const int tid    = threadIdx.x;
    const int wave   = tid >> 5;
    const int lane   = tid & 31;
    const int lane16 = lane & 15;
    const int hi     = (lane >> 4) & 1;

    const int bh = blockIdx.y;            // b*H + h
    const int b  = bh >> 4;
    const int h  = bh & 15;
    const int q0 = blockIdx.x * 128 + wave * 16;

    const _Float16* Qb = qkv + 0 * QKV_HALVES + (size_t)bh * S_ * HD_;
    const _Float16* Kb = qkv + 1 * QKV_HALVES + (size_t)bh * S_ * HD_;
    const _Float16* Vb = qkv + 2 * QKV_HALVES + (size_t)bh * S_ * HD_;

    // Q fragments (A-layout), hd chunks 0-31 and 32-63, held for whole loop.
    v16h qa[2];
#pragma unroll
    for (int c = 0; c < 2; ++c) {
        const _Float16* base = Qb + (size_t)(q0 + lane16) * HD_ + c * 32;
        qa[c] = ld2(base + (hi ? 16 : 0), base + (hi ? 24 : 8));
    }

    // all-ones B fragment: D = P @ ones puts rowsum(m) in every column,
    // so each lane's o4[r] directly tracks its row's softmax denominator.
    v16h ones;
#pragma unroll
    for (int i = 0; i < 16; ++i) ones[i] = (_Float16)1.0f;

    v8f o0 = {}, o1 = {}, o2 = {}, o3 = {}, o4 = {};
    float mrow[8];
#pragma unroll
    for (int r = 0; r < 8; ++r) mrow[r] = -INFINITY;

    _Float16* Pw = &Plds[wave * 16 * PSTR];
    const int skey = tid >> 3;            // staging: key row, 8 threads/row
    const int shb  = (tid & 7) * 8;       // staging: hd base

    for (int kb = 0; kb < S_; kb += 32) {
        // prefetch next chunk
        if (kb + 32 < S_) {
            __builtin_prefetch(Kb + (size_t)(kb + 32 + skey) * HD_ + shb, 0, 1);
            __builtin_prefetch(Vb + (size_t)(kb + 32 + skey) * HD_ + shb, 0, 1);
        }
        __syncthreads();
        // ---- stage K chunk [key][hd] via async-to-LDS DMA
        async_copy_b128(&Klds[skey * KSTR + shb],
                        Kb + (size_t)(kb + skey) * HD_ + shb);
        // ---- stage V chunk transposed [hd][key] (needs VGPR bounce)
        {
            v8h vg = *(const v8h*)(Vb + (size_t)(kb + skey) * HD_ + shb);
#pragma unroll
            for (int i = 0; i < 8; ++i)
                Vlds[(shb + i) * VSTR + skey] = vg[i];
        }
        wait_async0();
        __syncthreads();

        // ---- scores: S[16q x 32k] = Q @ K^T (contraction over hd=64)
        v8f s0 = {}, s1 = {};
#pragma unroll
        for (int c = 0; c < 2; ++c) {
            const _Float16* k0 = &Klds[(0 * 16 + lane16) * KSTR + c * 32 + hi * 16];
            const _Float16* k1 = &Klds[(1 * 16 + lane16) * KSTR + c * 32 + hi * 16];
            s0 = wmma_f16(qa[c], ld2(k0, k0 + 8), s0);
            s1 = wmma_f16(qa[c], ld2(k1, k1 + 8), s1);
        }

        // ---- online softmax: row max via half-wave xor shuffles only
        float nm[8], al[8], p0[8], p1[8];
#pragma unroll
        for (int r = 0; r < 8; ++r) {
            float x = fmaxf(s0[r], s1[r]);
#pragma unroll
            for (int msk = 8; msk > 0; msk >>= 1)
                x = fmaxf(x, __shfl_xor(x, msk, 32));
            nm[r] = fmaxf(mrow[r], x);
            al[r] = __expf(mrow[r] - nm[r]);
            p0[r] = __expf(s0[r] - nm[r]);
            p1[r] = __expf(s1[r] - nm[r]);
            mrow[r] = nm[r];
        }
#pragma unroll
        for (int r = 0; r < 8; ++r) {
            o0[r] *= al[r]; o1[r] *= al[r]; o2[r] *= al[r]; o3[r] *= al[r];
            o4[r] *= al[r];                    // running denominator rescale
        }

        // ---- P (C-layout regs) -> LDS [row][key]
#pragma unroll
        for (int r = 0; r < 8; ++r) {
            const int m = r + hi * 8;
            Pw[m * PSTR + lane16]      = (_Float16)p0[r];
            Pw[m * PSTR + 16 + lane16] = (_Float16)p1[r];
        }
        __syncthreads();

        // ---- O += P @ V, denominator += P @ ones
        const _Float16* prow = &Pw[lane16 * PSTR];
        v16h pa = ld2(prow + (hi ? 16 : 0), prow + (hi ? 24 : 8));
        const _Float16* v0p = &Vlds[(0 * 16 + lane16) * VSTR + hi * 16];
        const _Float16* v1p = &Vlds[(1 * 16 + lane16) * VSTR + hi * 16];
        const _Float16* v2p = &Vlds[(2 * 16 + lane16) * VSTR + hi * 16];
        const _Float16* v3p = &Vlds[(3 * 16 + lane16) * VSTR + hi * 16];
        o0 = wmma_f16(pa, ld2(v0p, v0p + 8), o0);
        o1 = wmma_f16(pa, ld2(v1p, v1p + 8), o1);
        o2 = wmma_f16(pa, ld2(v2p, v2p + 8), o2);
        o3 = wmma_f16(pa, ld2(v3p, v3p + 8), o3);
        o4 = wmma_f16(pa, ones, o4);
    }

    // ---- normalize + store attn out f16 as [B, S, E] (heads interleaved)
#pragma unroll
    for (int r = 0; r < 8; ++r) {
        const float inv = 1.0f / o4[r];        // rowsum lives in every column
        const int q = q0 + r + hi * 8;
        _Float16* dst = attnout + ((size_t)b * S_ + q) * E_ + h * HD_ + lane16;
        dst[0]  = (_Float16)(o0[r] * inv);
        dst[16] = (_Float16)(o1[r] * inv);
        dst[32] = (_Float16)(o2[r] * inv);
        dst[48] = (_Float16)(o3[r] * inv);
    }
}

// ---------------------------------------------------------------------------
extern "C" void kernel_launch(void* const* d_in, const int* in_sizes, int n_in,
                              void* d_out, int out_size, void* d_ws, size_t ws_size,
                              hipStream_t stream) {
    const float* q     = (const float*)d_in[0];   // [B,S,E]
    const float* w_in  = (const float*)d_in[1];   // [3E,E]
    const float* b_in  = (const float*)d_in[2];   // [3E]
    const float* w_out = (const float*)d_in[3];   // [E,E]
    const float* b_out = (const float*)d_in[4];   // [E]
    float* out = (float*)d_out;                   // [B,S,E] f32

    // workspace: f16 qkv (3 * 4M halves = 24 MiB) + f16 attn out (8 MiB)
    _Float16* qkv_ws  = (_Float16*)d_ws;
    _Float16* attn_ws = qkv_ws + 3 * QKV_HALVES;

    // 1) QKV projection: [4096,1024] x [3072,1024]^T
    gemm_wmma_kernel<true, 0><<<dim3(MROWS / 128, (3 * E_) / 32), 256, 0, stream>>>(
        (const void*)q, w_in, b_in, (void*)qkv_ws, MROWS, 3 * E_, E_);

    // 2) flash attention
    flash_attn_kernel<<<dim3(S_ / 128, B_ * H_), 256, 0, stream>>>(qkv_ws, attn_ws);

    // 3) output projection: [4096,1024] x [1024,1024]^T -> f32 d_out
    gemm_wmma_kernel<false, 1><<<dim3(MROWS / 128, E_ / 32), 256, 0, stream>>>(
        (const void*)attn_ws, w_out, b_out, (void*)out, MROWS, E_, E_);
}
